// SparseMoe_78752520340032
// MI455X (gfx1250) — compile-verified, matrix operand order checked by reading, not run on hardware
//
#include <hip/hip_runtime.h>
#include <hip/hip_bf16.h>

// ---------------- problem constants (match reference) ----------------
constexpr int TB    = 4;
constexpr int TS    = 2048;
constexpr int TT    = TB * TS;     // 8192 tokens
constexpr int D_IN  = 1024;
constexpr int D_OUT = 1024;
constexpr int HID   = 2048;
constexpr int NE    = 8;           // routed experts
constexpr int MT    = 32;          // token tile (2 WMMA M-fragments)

// dynamic LDS blob layout: h tile | x tile | token ids | gate weights
constexpr unsigned SMEM_H      = MT * HID * 2;           // 131072 B
constexpr unsigned SMEM_XT     = MT * D_IN * 2;          // 65536 B
constexpr unsigned SMEM_SHARED = SMEM_H + SMEM_XT;               // shared kernel
constexpr unsigned SMEM_ROUTED = SMEM_H + SMEM_XT + MT * 8;      // + tok/twt

// ---------------- WMMA / vector types ----------------
typedef __attribute__((ext_vector_type(16))) __bf16          v16bf;
typedef __attribute__((ext_vector_type(8)))  float           v8f;
typedef __attribute__((ext_vector_type(8)))  unsigned short  us8;
typedef __attribute__((ext_vector_type(16))) unsigned short  us16;
typedef __attribute__((ext_vector_type(4)))  unsigned int    uint4v;
typedef __attribute__((ext_vector_type(8)))  int             int8v;
typedef __attribute__((ext_vector_type(4)))  int             int4v;

static __device__ __forceinline__ v8f wmma_bf16(v16bf a, v16bf b, v8f c) {
    return __builtin_amdgcn_wmma_f32_16x16x32_bf16(
        /*neg_a=*/false, a, /*neg_b=*/false, b,
        /*c_mod=*/(short)0, c, /*reuse_a=*/false, /*reuse_b=*/false);
}

static __device__ __forceinline__ unsigned short f2bf(float f) {
    unsigned int u = __builtin_bit_cast(unsigned int, f);
    u += 0x7FFFu + ((u >> 16) & 1u);           // round-to-nearest-even
    return (unsigned short)(u >> 16);
}

// One 16-bit A/B fragment for v_wmma_*_16x16x32: lane (half = lane>>4) holds
// K = {k0+8h..+7} in elems 0..7 and K = {k0+16+8h..+7} in elems 8..15 of its
// row -> two contiguous 16-byte loads (global_load_b128 or ds_load_b128).
static __device__ __forceinline__ v16bf load_frag(const unsigned short* rowbase,
                                                  int k0, int half) {
    const us8* p = reinterpret_cast<const us8*>(rowbase + k0 + (half << 3));
    union { us16 v; us8 h[2]; } u;
    u.h[0] = p[0];
    u.h[1] = p[2];   // +16 elements
    return __builtin_bit_cast(v16bf, u.v);
}

// ---------------- conversion kernels ----------------
__global__ void cvt_bf16_kernel(const float* __restrict__ in,
                                unsigned short* __restrict__ out, int n) {
    int stride = gridDim.x * blockDim.x;
    for (int i = blockIdx.x * blockDim.x + threadIdx.x; i < n; i += stride)
        out[i] = f2bf(in[i]);
}

// [R,C] fp32 row-major -> [C,R] bf16 row-major, coalesced both sides.
// grid: (C/32, R/32, batch), block: (32, 8)
__global__ void cvt_transpose_kernel(const float* __restrict__ in,
                                     unsigned short* __restrict__ out,
                                     int R, int C) {
    __shared__ float tile[32][33];
    size_t slice = (size_t)R * C;
    in  += (size_t)blockIdx.z * slice;
    out += (size_t)blockIdx.z * slice;
    int c0 = blockIdx.x * 32, r0 = blockIdx.y * 32;
    int tx = threadIdx.x, ty = threadIdx.y;
    #pragma unroll
    for (int j = 0; j < 4; ++j)
        tile[ty + j * 8][tx] = in[(size_t)(r0 + ty + j * 8) * C + c0 + tx];
    __syncthreads();
    #pragma unroll
    for (int j = 0; j < 4; ++j)
        out[(size_t)(c0 + ty + j * 8) * R + r0 + tx] = f2bf(tile[tx][ty + j * 8]);
}

// ---------------- gating: softmax + top-2 + per-expert lists ----------------
__global__ void gate_topk_kernel(const float* __restrict__ x,
                                 const float* __restrict__ gw,
                                 const float* __restrict__ gb,
                                 float* __restrict__ ew_out,     // [T, E]
                                 int*   __restrict__ counts,     // [E]
                                 int*   __restrict__ idxs,       // [E, T]
                                 float* __restrict__ wgts) {     // [E, T]
    int wave = (blockIdx.x * blockDim.x + threadIdx.x) >> 5;
    int lane = threadIdx.x & 31;
    if (wave >= TT) return;
    const float* xr = x + (size_t)wave * D_IN;
    float acc[NE] = {};
    for (int i = lane; i < D_IN; i += 32) {
        float xv = xr[i];
        const float* g = gw + (size_t)i * NE;
        #pragma unroll
        for (int e = 0; e < NE; ++e) acc[e] += xv * g[e];
    }
    #pragma unroll
    for (int e = 0; e < NE; ++e)
        for (int off = 16; off > 0; off >>= 1)
            acc[e] += __shfl_down(acc[e], off, 32);
    if (lane == 0) {
        float lg[NE], mx = -3.4e38f;
        #pragma unroll
        for (int e = 0; e < NE; ++e) { lg[e] = acc[e] + gb[e]; mx = fmaxf(mx, lg[e]); }
        float sum = 0.f;
        #pragma unroll
        for (int e = 0; e < NE; ++e) { lg[e] = __expf(lg[e] - mx); sum += lg[e]; }
        float inv = 1.f / sum;
        #pragma unroll
        for (int e = 0; e < NE; ++e) { lg[e] *= inv; ew_out[(size_t)wave * NE + e] = lg[e]; }
        int i0 = 0;
        #pragma unroll
        for (int e = 1; e < NE; ++e) if (lg[e] > lg[i0]) i0 = e;
        int i1 = (i0 == 0) ? 1 : 0;
        #pragma unroll
        for (int e = 0; e < NE; ++e) if (e != i0 && lg[e] > lg[i1]) i1 = e;
        float s = lg[i0] + lg[i1];
        int p0 = atomicAdd(&counts[i0], 1);
        idxs[(size_t)i0 * TT + p0] = wave; wgts[(size_t)i0 * TT + p0] = lg[i0] / s;
        int p1 = atomicAdd(&counts[i1], 1);
        idxs[(size_t)i1 * TT + p1] = wave; wgts[(size_t)i1 * TT + p1] = lg[i1] / s;
    }
}

// ====================== expert MLP tile (M=32, two-phase) ==================
// x tile staged in LDS once per block (TDM for the contiguous case, coalesced
// gather for the routed case). Phase 1 A-fragments come from the DS pipe,
// B-fragments stream from global/L2. Per k-step each group does 8 WMMAs from
// 4 LDS A-loads + 8 global B-loads.

__global__ void __launch_bounds__(256)
shared_expert_kernel(const unsigned short* __restrict__ xb,    // [T, D_IN] bf16
                     const unsigned short* __restrict__ w1t,   // [HID, D_IN]
                     const unsigned short* __restrict__ w2t,   // [D_OUT, HID]
                     const float* __restrict__ sb1,            // [HID]
                     const float* __restrict__ sb2,            // [D_OUT]
                     float* __restrict__ out) {                // [T, D_OUT]
    extern __shared__ char smem[];
    unsigned short* hsh   = (unsigned short*)smem;             // [32][HID]
    unsigned short* xtile = (unsigned short*)(smem + SMEM_H);  // [32][D_IN]

    const int tile = blockIdx.x;
    const int wave = threadIdx.x >> 5;
    const int lane = threadIdx.x & 31;
    const int half = lane >> 4;
    const int lm   = lane & 15;
    const int t0   = tile * MT;

    // ---- stage x tile (32 x 1024 bf16 = 64 KB) into LDS via TDM ----
#if __has_builtin(__builtin_amdgcn_tensor_load_to_lds) && \
    __has_builtin(__builtin_amdgcn_s_wait_tensorcnt)
    if (threadIdx.x < 32) {                       // one wave issues the TDM op
        unsigned long long ga =
            (unsigned long long)(const void*)(xb + (size_t)t0 * D_IN);
        unsigned lds_off = (unsigned)(unsigned long long)(void*)xtile;
        uint4v g0;
        g0[0] = 1u;                                       // count=1, user mode
        g0[1] = lds_off;                                  // LDS dest (bytes)
        g0[2] = (unsigned)ga;                             // global_addr[31:0]
        g0[3] = (unsigned)((ga >> 32) & 0x1FFFFFFu) | (2u << 30);  // type=2
        int8v g1;
        g1[0] = (int)(1u << 16);                          // data_size: 2 bytes
        g1[1] = (int)((unsigned)(D_IN & 0xFFFF) << 16);   // tensor_dim0 lo
        g1[2] = (int)((unsigned)(D_IN >> 16) |
                      ((unsigned)(TT & 0xFFFF) << 16));   // td0 hi | td1 lo
        g1[3] = (int)((unsigned)(TT >> 16) |
                      ((unsigned)D_IN << 16));            // td1 hi | tile_dim0
        g1[4] = MT;                                       // tile_dim1 = 32
        g1[5] = D_IN;                                     // dim0_stride low
        g1[6] = 0;
        g1[7] = 0;
        int4v gz4 = {0, 0, 0, 0};
        int8v gz8 = {0, 0, 0, 0, 0, 0, 0, 0};
        __builtin_amdgcn_tensor_load_to_lds(g0, g1, gz4, gz4, gz8, 0);
        __builtin_amdgcn_s_wait_tensorcnt(0);
    }
#else
    for (int i = threadIdx.x; i < MT * (D_IN / 8); i += 256) {
        int row = i >> 7, c = i & 127;       // 128 x us8 chunks per row
        reinterpret_cast<us8*>(xtile + row * D_IN)[c] =
            reinterpret_cast<const us8*>(xb + (size_t)(t0 + row) * D_IN)[c];
    }
#endif
    __syncthreads();

    const unsigned short* arow0 = xtile + lm * D_IN;
    const unsigned short* arow1 = xtile + (16 + lm) * D_IN;

    // ---------- phase 1 ----------
    for (int g = 0; g < 4; ++g) {
        const int nbase = wave * 256 + g * 64;
        v8f acc[2][4];
        const unsigned short* brow[4];
        #pragma unroll
        for (int j = 0; j < 4; ++j) {
            float bias = sb1[nbase + j * 16 + lm];
            #pragma unroll
            for (int r = 0; r < 8; ++r) { acc[0][j][r] = bias; acc[1][j][r] = bias; }
            brow[j] = w1t + (size_t)(nbase + j * 16 + lm) * D_IN;
        }
        #pragma unroll 2
        for (int k0 = 0; k0 < D_IN; k0 += 32) {
            v16bf a0 = load_frag(arow0, k0, half);
            v16bf a1 = load_frag(arow1, k0, half);
            #pragma unroll
            for (int j = 0; j < 4; ++j) {
                v16bf b = load_frag(brow[j], k0, half);
                acc[0][j] = wmma_bf16(a0, b, acc[0][j]);
                acc[1][j] = wmma_bf16(a1, b, acc[1][j]);
            }
        }
        #pragma unroll
        for (int j = 0; j < 4; ++j)
            #pragma unroll
            for (int mi = 0; mi < 2; ++mi)
                #pragma unroll
                for (int r = 0; r < 8; ++r)
                    hsh[(mi * 16 + r + 8 * half) * HID + nbase + j * 16 + lm] =
                        f2bf(fmaxf(acc[mi][j][r], 0.f));
    }
    __syncthreads();

    // ---------- phase 2 ----------
    const unsigned short* hrow0 = hsh + (size_t)lm * HID;
    const unsigned short* hrow1 = hsh + (size_t)(16 + lm) * HID;
    for (int g = 0; g < 2; ++g) {
        const int nbase = wave * 128 + g * 64;
        v8f acc[2][4];
        const unsigned short* brow[4];
        #pragma unroll
        for (int j = 0; j < 4; ++j) {
            float bias = sb2[nbase + j * 16 + lm];
            #pragma unroll
            for (int r = 0; r < 8; ++r) { acc[0][j][r] = bias; acc[1][j][r] = bias; }
            brow[j] = w2t + (size_t)(nbase + j * 16 + lm) * HID;
        }
        #pragma unroll 2
        for (int k0 = 0; k0 < HID; k0 += 32) {
            v16bf a0 = load_frag(hrow0, k0, half);
            v16bf a1 = load_frag(hrow1, k0, half);
            #pragma unroll
            for (int j = 0; j < 4; ++j) {
                v16bf b = load_frag(brow[j], k0, half);
                acc[0][j] = wmma_bf16(a0, b, acc[0][j]);
                acc[1][j] = wmma_bf16(a1, b, acc[1][j]);
            }
        }
        #pragma unroll
        for (int j = 0; j < 4; ++j)
            #pragma unroll
            for (int mi = 0; mi < 2; ++mi)
                #pragma unroll
                for (int r = 0; r < 8; ++r)
                    out[(size_t)(t0 + mi * 16 + r + 8 * half) * D_OUT +
                        nbase + j * 16 + lm] = acc[mi][j][r];
    }
}

__global__ void __launch_bounds__(256)
routed_expert_kernel(const unsigned short* __restrict__ xb,
                     const unsigned short* __restrict__ w1t,   // [E][HID][D_IN]
                     const unsigned short* __restrict__ w2t,   // [E][D_OUT][HID]
                     const float* __restrict__ eb1,            // [E, HID]
                     const float* __restrict__ eb2,            // [E, D_OUT]
                     const int*   __restrict__ counts,
                     const int*   __restrict__ idxs,
                     const float* __restrict__ wgts,
                     float* __restrict__ out) {
    const int e    = blockIdx.y;
    const int tile = blockIdx.x;
    const int cnt  = counts[e];
    if (tile * MT >= cnt) return;                  // uniform per block

    extern __shared__ char smem[];
    unsigned short* hsh   = (unsigned short*)smem;                 // [32][HID]
    unsigned short* xtile = (unsigned short*)(smem + SMEM_H);      // [32][D_IN]
    int*   tok = (int*)(smem + SMEM_H + SMEM_XT);                  // [32]
    float* twt = (float*)(smem + SMEM_H + SMEM_XT + MT * 4);       // [32]

    const int wave = threadIdx.x >> 5;
    const int lane = threadIdx.x & 31;
    const int half = lane >> 4;
    const int lm   = lane & 15;

    if (threadIdx.x < MT) {
        int row = tile * MT + threadIdx.x;
        if (row < cnt) { tok[threadIdx.x] = idxs[(size_t)e * TT + row];
                         twt[threadIdx.x] = wgts[(size_t)e * TT + row]; }
        else           { tok[threadIdx.x] = 0; twt[threadIdx.x] = 0.f; }
    }
    __syncthreads();

    // gather-stage x rows into LDS (one read per gathered row per block)
    for (int i = threadIdx.x; i < MT * (D_IN / 8); i += 256) {
        int row = i >> 7, c = i & 127;
        reinterpret_cast<us8*>(xtile + row * D_IN)[c] =
            reinterpret_cast<const us8*>(xb + (size_t)tok[row] * D_IN)[c];
    }
    __syncthreads();

    const unsigned short* W1 = w1t + (size_t)e * HID * D_IN;
    const unsigned short* W2 = w2t + (size_t)e * D_OUT * HID;
    const unsigned short* arow0 = xtile + lm * D_IN;
    const unsigned short* arow1 = xtile + (16 + lm) * D_IN;

    // ---------- phase 1 ----------
    for (int g = 0; g < 4; ++g) {
        const int nbase = wave * 256 + g * 64;
        v8f acc[2][4];
        const unsigned short* brow[4];
        #pragma unroll
        for (int j = 0; j < 4; ++j) {
            float bias = eb1[(size_t)e * HID + nbase + j * 16 + lm];
            #pragma unroll
            for (int r = 0; r < 8; ++r) { acc[0][j][r] = bias; acc[1][j][r] = bias; }
            brow[j] = W1 + (size_t)(nbase + j * 16 + lm) * D_IN;
        }
        #pragma unroll 2
        for (int k0 = 0; k0 < D_IN; k0 += 32) {
            v16bf a0 = load_frag(arow0, k0, half);
            v16bf a1 = load_frag(arow1, k0, half);
            #pragma unroll
            for (int j = 0; j < 4; ++j) {
                v16bf b = load_frag(brow[j], k0, half);
                acc[0][j] = wmma_bf16(a0, b, acc[0][j]);
                acc[1][j] = wmma_bf16(a1, b, acc[1][j]);
            }
        }
        #pragma unroll
        for (int j = 0; j < 4; ++j)
            #pragma unroll
            for (int mi = 0; mi < 2; ++mi)
                #pragma unroll
                for (int r = 0; r < 8; ++r)
                    hsh[(mi * 16 + r + 8 * half) * HID + nbase + j * 16 + lm] =
                        f2bf(fmaxf(acc[mi][j][r], 0.f));
    }
    __syncthreads();

    // ---------- phase 2 (scale by gate weight, eb2 folded in first) --------
    const unsigned short* hrow0 = hsh + (size_t)lm * HID;
    const unsigned short* hrow1 = hsh + (size_t)(16 + lm) * HID;
    for (int g = 0; g < 2; ++g) {
        const int nbase = wave * 128 + g * 64;
        v8f acc[2][4];
        const unsigned short* brow[4];
        #pragma unroll
        for (int j = 0; j < 4; ++j) {
            float bias = eb2[(size_t)e * D_OUT + nbase + j * 16 + lm];
            #pragma unroll
            for (int r = 0; r < 8; ++r) { acc[0][j][r] = bias; acc[1][j][r] = bias; }
            brow[j] = W2 + (size_t)(nbase + j * 16 + lm) * HID;
        }
        #pragma unroll 2
        for (int k0 = 0; k0 < HID; k0 += 32) {
            v16bf a0 = load_frag(hrow0, k0, half);
            v16bf a1 = load_frag(hrow1, k0, half);
            #pragma unroll
            for (int j = 0; j < 4; ++j) {
                v16bf b = load_frag(brow[j], k0, half);
                acc[0][j] = wmma_bf16(a0, b, acc[0][j]);
                acc[1][j] = wmma_bf16(a1, b, acc[1][j]);
            }
        }
        #pragma unroll
        for (int j = 0; j < 4; ++j)
            #pragma unroll
            for (int mi = 0; mi < 2; ++mi)
                #pragma unroll
                for (int r = 0; r < 8; ++r) {
                    int mrow = mi * 16 + r + 8 * half;
                    float v = acc[mi][j][r] * twt[mrow];
                    unsafeAtomicAdd(&out[(size_t)tok[mrow] * D_OUT +
                                         nbase + j * 16 + lm], v);
                }
    }
}

// ---------------- host launcher ----------------
extern "C" void kernel_launch(void* const* d_in, const int* in_sizes, int n_in,
                              void* d_out, int out_size, void* d_ws, size_t ws_size,
                              hipStream_t stream) {
    const float* x      = (const float*)d_in[0];
    const float* gate_w = (const float*)d_in[1];
    const float* gate_b = (const float*)d_in[2];
    const float* ew1    = (const float*)d_in[3];
    const float* eb1    = (const float*)d_in[4];
    const float* ew2    = (const float*)d_in[5];
    const float* eb2    = (const float*)d_in[6];
    const float* sw1    = (const float*)d_in[7];
    const float* sb1    = (const float*)d_in[8];
    const float* sw2    = (const float*)d_in[9];
    const float* sb2    = (const float*)d_in[10];

    float* out    = (float*)d_out;                      // [T, D_OUT]
    float* ew_out = out + (size_t)TT * D_OUT;           // [T, E]

    // workspace carve-up
    char* p = (char*)d_ws;
    unsigned short* xb   = (unsigned short*)p; p += (size_t)TT * D_IN * 2;
    unsigned short* w1t  = (unsigned short*)p; p += (size_t)NE * HID * D_IN * 2;
    unsigned short* w2t  = (unsigned short*)p; p += (size_t)NE * D_OUT * HID * 2;
    unsigned short* sw1t = (unsigned short*)p; p += (size_t)HID * D_IN * 2;
    unsigned short* sw2t = (unsigned short*)p; p += (size_t)D_OUT * HID * 2;
    int*   counts = (int*)p;   p += 256;                // padded
    int*   idxs   = (int*)p;   p += (size_t)NE * TT * 4;
    float* wgts   = (float*)p; p += (size_t)NE * TT * 4;

    (void)hipMemsetAsync(counts, 0, NE * sizeof(int), stream);

    cvt_bf16_kernel<<<4096, 256, 0, stream>>>(x, xb, TT * D_IN);
    cvt_transpose_kernel<<<dim3(HID / 32, D_IN / 32, NE), dim3(32, 8), 0, stream>>>(
        ew1, w1t, D_IN, HID);
    cvt_transpose_kernel<<<dim3(D_OUT / 32, HID / 32, NE), dim3(32, 8), 0, stream>>>(
        ew2, w2t, HID, D_OUT);
    cvt_transpose_kernel<<<dim3(HID / 32, D_IN / 32, 1), dim3(32, 8), 0, stream>>>(
        sw1, sw1t, D_IN, HID);
    cvt_transpose_kernel<<<dim3(D_OUT / 32, HID / 32, 1), dim3(32, 8), 0, stream>>>(
        sw2, sw2t, HID, D_OUT);

    gate_topk_kernel<<<TT / 8, 256, 0, stream>>>(x, gate_w, gate_b, ew_out,
                                                 counts, idxs, wgts);

    shared_expert_kernel<<<TT / MT, 256, SMEM_SHARED, stream>>>(
        xb, sw1t, sw2t, sb1, sb2, out);

    routed_expert_kernel<<<dim3(TT / MT, NE), 256, SMEM_ROUTED, stream>>>(
        xb, w1t, w2t, eb1, eb2, counts, idxs, wgts, out);
}
